// MoELayer_8177617731883
// MI455X (gfx1250) — compile-verified, matrix operand order checked by reading, not run on hardware
//
#include <hip/hip_runtime.h>

// ---------------------------------------------------------------------------
// MoE (top-2 of 8 experts), D=1024, H=2048, N tokens = 8192.
// bf16 WMMA path; M=32 token tiles for 2x B-fragment reuse (L2-BW bound);
// async global->LDS staging of the x tile (ASYNCcnt); weights folded into
// the activation so the output accumulates linearly across experts.
// ---------------------------------------------------------------------------

typedef __attribute__((ext_vector_type(16))) __bf16 v16bf;
typedef __attribute__((ext_vector_type(8)))  float  v8f;

#define N_EXPERTS 8
#define D_EMBD    1024
#define H_DIM     2048
#define MTILE     32     // tokens per block (two 16-row WMMA M-tiles)
#define KCHUNK    128    // expert-dim columns per fused chunk (16 per wave)
#define THREADS   256    // 8 waves (wave32)

__device__ __forceinline__ unsigned short f2bf_rne(float f) {
    unsigned int u = __float_as_uint(f);
    unsigned int r = u + 0x7FFFu + ((u >> 16) & 1u);   // round-to-nearest-even
    return (unsigned short)(r >> 16);
}

__device__ __forceinline__ v8f zero8() {
    v8f z = {0.f, 0.f, 0.f, 0.f, 0.f, 0.f, 0.f, 0.f};
    return z;
}

// Load a 16x32 bf16 A/B fragment from a row-major 16-row slab with leading
// dimension `ld`, K offset `kbase`.  Per ISA 16-bit layout: lane L holds row
// (L&15); lanes 0-15 carry K = kbase+{0..7, 16..23}, lanes 16-31 carry
// K = kbase+{8..15, 24..31}.  Two 16-byte loads per lane.
__device__ __forceinline__ v16bf load_frag16(const unsigned short* __restrict__ base,
                                             int ld, int kbase, int lane) {
    const int r  = lane & 15;
    const int h8 = (lane >> 4) * 8;
    const unsigned short* p = base + r * ld + kbase + h8;
    union { uint4 q[2]; v16bf v; } u;
    u.q[0] = *(const uint4*)(p);
    u.q[1] = *(const uint4*)(p + 16);
    return u.v;
}

// ---------------------------------------------------------------------------
// Kernel 1: f32 -> bf16 conversion (vectorized 4 elems/thread).
// ---------------------------------------------------------------------------
__global__ __launch_bounds__(256) void cvt_f32_to_bf16(const float* __restrict__ s,
                                                       unsigned short* __restrict__ d,
                                                       int n) {
    int i = (blockIdx.x * 256 + threadIdx.x) * 4;
    if (i + 3 < n) {
        float4 v = *(const float4*)(s + i);
        ushort4 o;
        o.x = f2bf_rne(v.x); o.y = f2bf_rne(v.y);
        o.z = f2bf_rne(v.z); o.w = f2bf_rne(v.w);
        *(ushort4*)(d + i) = o;
    }
}

// ---------------------------------------------------------------------------
// Kernel 2: router — logits, softmax, top-2, normalized combine weights [N,8].
// One wave32 per token.
// ---------------------------------------------------------------------------
__global__ __launch_bounds__(256) void router_topk(const float* __restrict__ x,
                                                   const float* __restrict__ rw,
                                                   float* __restrict__ wts,
                                                   int ntok) {
    const int wave = threadIdx.x >> 5;
    const int lane = threadIdx.x & 31;
    const int tok  = blockIdx.x * 8 + wave;
    if (tok >= ntok) return;

    const float* xr = x + (size_t)tok * D_EMBD;
    float acc[N_EXPERTS] = {0.f, 0.f, 0.f, 0.f, 0.f, 0.f, 0.f, 0.f};
    for (int d = lane; d < D_EMBD; d += 32) {
        float xv = xr[d];
#pragma unroll
        for (int e = 0; e < N_EXPERTS; ++e)
            acc[e] += xv * rw[e * D_EMBD + d];
    }
#pragma unroll
    for (int off = 16; off > 0; off >>= 1) {
#pragma unroll
        for (int e = 0; e < N_EXPERTS; ++e)
            acc[e] += __shfl_xor(acc[e], off, 32);
    }
    float mx = acc[0];
#pragma unroll
    for (int e = 1; e < N_EXPERTS; ++e) mx = fmaxf(mx, acc[e]);
    float p[N_EXPERTS], ssum = 0.f;
#pragma unroll
    for (int e = 0; e < N_EXPERTS; ++e) { p[e] = __expf(acc[e] - mx); ssum += p[e]; }
    float inv = 1.0f / ssum;
#pragma unroll
    for (int e = 0; e < N_EXPERTS; ++e) p[e] *= inv;
    int i1 = 0;
#pragma unroll
    for (int e = 1; e < N_EXPERTS; ++e) if (p[e] > p[i1]) i1 = e;
    int i2 = (i1 == 0) ? 1 : 0;
#pragma unroll
    for (int e = 0; e < N_EXPERTS; ++e) if (e != i1 && p[e] > p[i2]) i2 = e;
    float norm = 1.0f / (p[i1] + p[i2]);
    if (lane < N_EXPERTS) {
        float w = (lane == i1) ? p[i1] * norm : (lane == i2) ? p[i2] * norm : 0.0f;
        wts[(size_t)tok * N_EXPERTS + lane] = w;
    }
}

// ---------------------------------------------------------------------------
// Kernel 3: fused expert compute.  Block = 32 tokens x full D output, 8 waves.
// ---------------------------------------------------------------------------
__global__ __launch_bounds__(THREADS) void moe_expert_wmma(
        const unsigned short* __restrict__ xb,
        const unsigned short* __restrict__ f1b,
        const unsigned short* __restrict__ f2b,
        const float* __restrict__ wts,
        float* __restrict__ out) {
    __shared__ unsigned short xs[MTILE * D_EMBD];   // 64 KB: x tile, bf16
    __shared__ unsigned short hs[MTILE * KCHUNK];   //  8 KB: activated h chunk

    const int tid   = threadIdx.x;
    const int wave  = tid >> 5;
    const int lane  = tid & 31;
    const int tile0 = blockIdx.x * MTILE;

    // Stage x tile (32 x 1024 bf16 = 64 KB) into LDS with async b128 copies
    // (tracked by ASYNCcnt, no VGPR round-trip).
    {
        const char* gsrc = (const char*)(xb + (size_t)tile0 * D_EMBD);
        unsigned lds_base = (unsigned)(size_t)(&xs[0]);
        const int total = MTILE * D_EMBD * 2 / 16;  // 4096 x 16B chunks
        for (int i = tid; i < total; i += THREADS) {
            unsigned lds_addr = lds_base + (unsigned)i * 16u;
            unsigned long long gaddr = (unsigned long long)(size_t)(gsrc + (size_t)i * 16);
            asm volatile("global_load_async_to_lds_b128 %0, %1, off"
                         :: "v"(lds_addr), "v"(gaddr)
                         : "memory");
        }
        asm volatile("s_wait_asynccnt 0" ::: "memory");
    }
    __syncthreads();

    const int mbase = (lane < 16) ? 0 : 8;          // C/D layout: token half
    const int nlane = lane & 15;

    v8f acc0[8], acc1[8];   // out cols [wave*128, +128) for M-tiles 0 and 1
#pragma unroll
    for (int t = 0; t < 8; ++t) { acc0[t] = zero8(); acc1[t] = zero8(); }

    for (int e = 0; e < N_EXPERTS; ++e) {
        // Per-token combine weights for this lane's C-layout rows.
        float wrow0[8], wrow1[8];
        float asum = 0.f;
#pragma unroll
        for (int r = 0; r < 8; ++r) {
            wrow0[r] = wts[(size_t)(tile0 + mbase + r) * N_EXPERTS + e];
            wrow1[r] = wts[(size_t)(tile0 + 16 + mbase + r) * N_EXPERTS + e];
            asum += fabsf(wrow0[r]) + fabsf(wrow1[r]);
        }
        // Uniform skip if no token in this tile routed to expert e.
        if (__ballot(asum != 0.0f) == 0) continue;

        const unsigned short* f1e = f1b + (size_t)e * H_DIM * D_EMBD;
        const unsigned short* f2e = f2b + (size_t)e * D_EMBD * H_DIM;

        for (int kc = 0; kc < H_DIM; kc += KCHUNK) {
            // ---- GEMM1: h (32 tokens x 16 cols per wave), K = 1024.
            //      One B fragment feeds two M-tiles (2x L2 reuse). ----
            v8f hacc0 = zero8(), hacc1 = zero8();
            const unsigned short* f1slab = f1e + (size_t)(kc + wave * 16) * D_EMBD;
#pragma unroll 2
            for (int k = 0; k < D_EMBD; k += 32) {
                __builtin_prefetch(f1slab + (size_t)nlane * D_EMBD + k + 128, 0, 0);
                v16bf b  = load_frag16(f1slab, D_EMBD, k, lane);
                v16bf a0 = load_frag16(xs, D_EMBD, k, lane);
                v16bf a1 = load_frag16(xs + 16 * D_EMBD, D_EMBD, k, lane);
                hacc0 = __builtin_amdgcn_wmma_f32_16x16x32_bf16(
                        false, a0, false, b, (short)0, hacc0, false, false);
                hacc1 = __builtin_amdgcn_wmma_f32_16x16x32_bf16(
                        false, a1, false, b, (short)0, hacc1, false, false);
            }
            // Activation relu(v)^2 with combine weight folded in; store bf16.
            unsigned short* hdst = hs + wave * 16 + nlane;
#pragma unroll
            for (int r = 0; r < 8; ++r) {
                float v0 = fmaxf(hacc0[r], 0.0f);
                float v1 = fmaxf(hacc1[r], 0.0f);
                v0 = wrow0[r] * v0 * v0;
                v1 = wrow1[r] * v1 * v1;
                hdst[(mbase + r) * KCHUNK]      = f2bf_rne(v0);
                hdst[(16 + mbase + r) * KCHUNK] = f2bf_rne(v1);
            }
            __syncthreads();

            // ---- GEMM2: out[32 x 1024] += h[32 x 128] @ fc2^T chunk.
            //      Each B fragment feeds two M-tiles. ----
#pragma unroll
            for (int kk = 0; kk < KCHUNK; kk += 32) {
                v16bf a0 = load_frag16(hs, KCHUNK, kk, lane);
                v16bf a1 = load_frag16(hs + 16 * KCHUNK, KCHUNK, kk, lane);
#pragma unroll
                for (int t = 0; t < 8; ++t) {
                    const unsigned short* f2slab =
                        f2e + (size_t)(wave * 128 + t * 16) * H_DIM;
                    v16bf b2 = load_frag16(f2slab, H_DIM, kc + kk, lane);
                    acc0[t] = __builtin_amdgcn_wmma_f32_16x16x32_bf16(
                            false, a0, false, b2, (short)0, acc0[t], false, false);
                    acc1[t] = __builtin_amdgcn_wmma_f32_16x16x32_bf16(
                            false, a1, false, b2, (short)0, acc1[t], false, false);
                }
            }
            __syncthreads();   // hs reused next chunk
        }
    }

    // Epilogue: single store per output element (weights already folded in).
#pragma unroll
    for (int t = 0; t < 8; ++t) {
        int col = wave * 128 + t * 16 + nlane;
#pragma unroll
        for (int r = 0; r < 8; ++r) {
            out[(size_t)(tile0 + mbase + r) * D_EMBD + col]      = acc0[t][r];
            out[(size_t)(tile0 + 16 + mbase + r) * D_EMBD + col] = acc1[t][r];
        }
    }
}

// ---------------------------------------------------------------------------
// Host launcher.
// ---------------------------------------------------------------------------
extern "C" void kernel_launch(void* const* d_in, const int* in_sizes, int n_in,
                              void* d_out, int out_size, void* d_ws, size_t ws_size,
                              hipStream_t stream) {
    (void)n_in; (void)out_size; (void)ws_size;
    const float* x        = (const float*)d_in[0];
    const float* router_w = (const float*)d_in[1];
    const float* fc1_w    = (const float*)d_in[2];
    const float* fc2_w    = (const float*)d_in[3];
    float* out            = (float*)d_out;

    const size_t nx  = (size_t)in_sizes[0];   // N * 1024
    const size_t nf1 = (size_t)in_sizes[2];   // 8 * 2048 * 1024
    const size_t nf2 = (size_t)in_sizes[3];   // 8 * 1024 * 2048
    const int ntok = (int)(nx / D_EMBD);      // 8192

    // Workspace layout: [wts f32 N*8][x bf16][fc1 bf16][fc2 bf16]
    char* w = (char*)d_ws;
    float* wts = (float*)w;
    size_t off = ((size_t)ntok * N_EXPERTS * sizeof(float) + 255) & ~(size_t)255;
    unsigned short* xb  = (unsigned short*)(w + off);
    unsigned short* f1b = xb + nx;
    unsigned short* f2b = f1b + nf1;

    cvt_f32_to_bf16<<<dim3((unsigned)(nx  / 1024)), dim3(256), 0, stream>>>(x,     xb,  (int)nx);
    cvt_f32_to_bf16<<<dim3((unsigned)(nf1 / 1024)), dim3(256), 0, stream>>>(fc1_w, f1b, (int)nf1);
    cvt_f32_to_bf16<<<dim3((unsigned)(nf2 / 1024)), dim3(256), 0, stream>>>(fc2_w, f2b, (int)nf2);

    router_topk<<<dim3((unsigned)((ntok + 7) / 8)), dim3(256), 0, stream>>>(
        x, router_w, wts, ntok);

    moe_expert_wmma<<<dim3((unsigned)(ntok / MTILE)), dim3(THREADS), 0, stream>>>(
        xb, f1b, f2b, wts, out);
}
